// PsRoiOffset_41755672051998
// MI455X (gfx1250) — matrix operand order repaired and checked by reading, not compiled, not run on hardware
//
#include <hip/hip_runtime.h>
#include <stdint.h>

#define KPOOL     7
#define C_OUT     10
#define HF        200
#define WF        200
#define C_IN      490          // KPOOL*KPOOL*C_OUT
#define C_IN_PAD  512
#define N_OFF     98           // 2*KPOOL*KPOOL
#define N_PAD     112          // 7 WMMA N-tiles
#define K_TOTAL   (9 * C_IN_PAD)   // 4608
#define N_ROIS    512
#define N_PIX     (HF * WF)    // 40000
#define FSTRIDE   16.0f
#define LAMDA     0.1f

// LDS staging: per (tap, channel-half) slice of B: 112 rows x 256 ch bf16,
// row padded by 8 elements so ds_load_b128 across lanes is conflict-free.
#define ROW_ELEMS 264          // 256 + 8 pad
#define LDS_ELEMS (N_PAD * ROW_ELEMS)   // 29568 -> 59136 bytes

typedef __attribute__((ext_vector_type(16))) __bf16        v16bf;
typedef __attribute__((ext_vector_type(8)))  float         v8f;
typedef __attribute__((ext_vector_type(8)))  unsigned int  v8u;
typedef __attribute__((ext_vector_type(4)))  int           v4i;

#if __has_builtin(__builtin_amdgcn_global_load_async_to_lds_b128)
typedef __attribute__((address_space(1))) v4i v4i_as1;
typedef __attribute__((address_space(3))) v4i v4i_as3;
#endif

union Frag { v8u u; v16bf b; };

__device__ __forceinline__ unsigned short f2bf(float f) {
  unsigned int u = __float_as_uint(f);
  u += 0x7FFFu + ((u >> 16) & 1u);   // round-to-nearest-even
  return (unsigned short)(u >> 16);
}

// ---------------- pack features f32 -> bf16, channel-padded to 512 ----------
__global__ void pack_feat(const float* __restrict__ feat,
                          unsigned short* __restrict__ out) {
  int idx = blockIdx.x * blockDim.x + threadIdx.x;   // over N_PIX * C_IN_PAD
  if (idx >= N_PIX * C_IN_PAD) return;
  int c = idx & (C_IN_PAD - 1);
  int p = idx >> 9;
  float v = (c < C_IN) ? feat[(size_t)p * C_IN + c] : 0.0f;
  out[idx] = f2bf(v);
}

// ------------- pack conv weights [3,3,490,98] -> bf16 [112][9*512] ----------
__global__ void pack_w(const float* __restrict__ w,
                       unsigned short* __restrict__ out) {
  int idx = blockIdx.x * blockDim.x + threadIdx.x;   // over N_PAD * K_TOTAL
  if (idx >= N_PAD * K_TOTAL) return;
  int k = idx % K_TOTAL;
  int n = idx / K_TOTAL;
  int t  = k >> 9;        // tap 0..8
  int ci = k & 511;       // padded input channel
  float v = 0.0f;
  if (n < N_OFF && ci < C_IN)
    v = w[((size_t)t * C_IN + ci) * N_OFF + n];
  out[idx] = f2bf(v);
}

// ---------------- implicit-GEMM 3x3 conv via bf16 WMMA ----------------------
// 10 waves/block; each wave computes TWO 16-pixel M-tiles x 112 channels.
// Weights for the current (tap, channel-half) are staged in LDS by the whole
// block (async global->LDS when available) and read back with ds_load_b128.
__global__ void __launch_bounds__(320)
conv_wmma(const unsigned short* __restrict__ featb,
          const unsigned short* __restrict__ wb,
          float* __restrict__ offmap) {
  __shared__ unsigned short sh[LDS_ELEMS];

  const int lane = threadIdx.x & 31;
  const int wave = threadIdx.x >> 5;
  const int job  = blockIdx.x * 10 + wave;   // 0..1249, exact coverage
  const int tile0 = job * 2;                 // two M-tiles per wave
  const int m  = lane & 15;                  // A row / B column / D column
  const int kh = lane >> 4;                  // which K half this lane holds

  const int p0 = tile0 * 16 + m;
  const int p1 = p0 + 16;
  const int y0 = p0 / WF, x0 = p0 % WF;
  const int y1 = p1 / WF, x1 = p1 % WF;

  v8f acc[2][7] = {};

  for (int t = 0; t < 9; ++t) {
    const int dy = t / 3 - 1, dx = t % 3 - 1;
    const int yy0 = y0 + dy, xx0 = x0 + dx;
    const int yy1 = y1 + dy, xx1 = x1 + dx;
    const bool ok0 = (yy0 >= 0) && (yy0 < HF) && (xx0 >= 0) && (xx0 < WF);
    const bool ok1 = (yy1 >= 0) && (yy1 < HF) && (xx1 >= 0) && (xx1 < WF);
    const uint4* ab0 =
        (const uint4*)(featb + (ok0 ? (size_t)(yy0 * WF + xx0) * C_IN_PAD : 0));
    const uint4* ab1 =
        (const uint4*)(featb + (ok1 ? (size_t)(yy1 * WF + xx1) * C_IN_PAD : 0));

    for (int h = 0; h < 2; ++h) {            // channel halves (256 each)
      __syncthreads();                       // previous-stage LDS reads done
      // ---- stage B[tap t, channels h*256..h*256+255] into LDS -------------
      for (int c = threadIdx.x; c < N_PAD * 32; c += 320) {
        const int row = c >> 5, col = c & 31;
        const unsigned short* g =
            wb + (size_t)row * K_TOTAL + t * 512 + h * 256 + col * 8;
        unsigned short* l = sh + row * ROW_ELEMS + col * 8;
#if __has_builtin(__builtin_amdgcn_global_load_async_to_lds_b128)
        __builtin_amdgcn_global_load_async_to_lds_b128(
            (v4i_as1*)(void*)const_cast<unsigned short*>(g),
            (v4i_as3*)(void*)l, 0, 0);
#else
        *(uint4*)l = *(const uint4*)g;
#endif
      }
#if __has_builtin(__builtin_amdgcn_global_load_async_to_lds_b128)
#if __has_builtin(__builtin_amdgcn_s_wait_asynccnt)
      __builtin_amdgcn_s_wait_asynccnt(0);
#else
      asm volatile("s_wait_asynccnt 0" ::: "memory");
#endif
#endif
      __syncthreads();                       // staged weights visible

      // ---- compute: 8 x (K=32) chunks against staged weights --------------
      for (int ccl = 0; ccl < 8; ++ccl) {
        const int cc = h * 8 + ccl;
        // A fragments (16-bit A layout): lane(kh) V0-3 = K[kh*8..+8],
        // V4-7 = K[16+kh*8..+8]; zero at image borders.
        Frag A0; A0.u = (v8u)0u;
        if (ok0) {
          uint4 lo = ab0[cc * 4 + kh];
          uint4 hi = ab0[cc * 4 + 2 + kh];
          A0.u[0] = lo.x; A0.u[1] = lo.y; A0.u[2] = lo.z; A0.u[3] = lo.w;
          A0.u[4] = hi.x; A0.u[5] = hi.y; A0.u[6] = hi.z; A0.u[7] = hi.w;
        }
        Frag A1; A1.u = (v8u)0u;
        if (ok1) {
          uint4 lo = ab1[cc * 4 + kh];
          uint4 hi = ab1[cc * 4 + 2 + kh];
          A1.u[0] = lo.x; A1.u[1] = lo.y; A1.u[2] = lo.z; A1.u[3] = lo.w;
          A1.u[4] = hi.x; A1.u[5] = hi.y; A1.u[6] = hi.z; A1.u[7] = hi.w;
        }
#pragma unroll
        for (int nt = 0; nt < 7; ++nt) {
          // B fragment from LDS: lane = column n, V0-3 = K[kh*8..+8], V4-7 = K[16+kh*8..+8]
          const uint4* bb = (const uint4*)(sh + (size_t)(nt * 16 + m) * ROW_ELEMS
                                              + ccl * 32 + kh * 8);
          uint4 blo = bb[0];
          uint4 bhi = bb[2];
          Frag B;
          B.u[0] = blo.x; B.u[1] = blo.y; B.u[2] = blo.z; B.u[3] = blo.w;
          B.u[4] = bhi.x; B.u[5] = bhi.y; B.u[6] = bhi.z; B.u[7] = bhi.w;
          acc[0][nt] = __builtin_amdgcn_wmma_f32_16x16x32_bf16(
              false, A0.b, false, B.b, (short)0, acc[0][nt], false, false);
          acc[1][nt] = __builtin_amdgcn_wmma_f32_16x16x32_bf16(
              false, A1.b, false, B.b, (short)0, acc[1][nt], false, false);
        }
      }
    }
  }

  // D layout: VGPR i -> M = i + kh*8, N = lane&15
#pragma unroll
  for (int tt = 0; tt < 2; ++tt) {
#pragma unroll
    for (int nt = 0; nt < 7; ++nt) {
#pragma unroll
      for (int i = 0; i < 8; ++i) {
        int pix = (tile0 + tt) * 16 + kh * 8 + i;
        int ch  = nt * 16 + m;
        offmap[(size_t)pix * N_PAD + ch] = acc[tt][nt][i];
      }
    }
  }
}

// ---------------- deformable PS-RoI pooling (both stages) -------------------
__device__ __forceinline__ float bilin(const float* __restrict__ f, int stride,
                                       int ch, float y, float x) {
  y = fminf(fmaxf(y, 0.0f), (float)(HF - 1));
  x = fminf(fmaxf(x, 0.0f), (float)(WF - 1));
  float y0f = floorf(y), x0f = floorf(x);
  int y0 = (int)y0f, x0 = (int)x0f;
  int y1 = min(y0 + 1, HF - 1), x1 = min(x0 + 1, WF - 1);
  float wy = y - y0f, wx = x - x0f;
  float v00 = f[(size_t)(y0 * WF + x0) * stride + ch];
  float v01 = f[(size_t)(y0 * WF + x1) * stride + ch];
  float v10 = f[(size_t)(y1 * WF + x0) * stride + ch];
  float v11 = f[(size_t)(y1 * WF + x1) * stride + ch];
  return v00 * (1.0f - wy) * (1.0f - wx) + v01 * (1.0f - wy) * wx +
         v10 * wy * (1.0f - wx) + v11 * wy * wx;
}

__global__ void __launch_bounds__(512)
roi_pool(const float* __restrict__ feat, const float* __restrict__ offmap,
         const float* __restrict__ rois, float* __restrict__ out) {
  __shared__ float offs[KPOOL * KPOOL][2];
  const int r = blockIdx.x;
  const float x1  = rois[r * 5 + 1];
  const float y1r = rois[r * 5 + 2];
  const float x2  = rois[r * 5 + 3];
  const float y2  = rois[r * 5 + 4];
  const float rw = x2 - x1 + 1.0f, rh = y2 - y1r + 1.0f;
  const float fx1 = x1 / FSTRIDE, fy1 = y1r / FSTRIDE;
  const float bw = rw / FSTRIDE / KPOOL, bh = rh / FSTRIDE / KPOOL;
  const int tid = threadIdx.x;

  // stage 1: PS-RoI pool the offset map -> per-bin (dx, dy)
  if (tid < 2 * KPOOL * KPOOL) {
    int bin = tid >> 1, c = tid & 1;
    int bi = bin / KPOOL, bj = bin % KPOOL;
    float s = 0.0f;
#pragma unroll
    for (int sy = 0; sy < 2; ++sy)
#pragma unroll
      for (int sx = 0; sx < 2; ++sx) {
        float gx = fx1 + ((float)bj + ((float)sx + 0.5f) * 0.5f) * bw;
        float gy = fy1 + ((float)bi + ((float)sy + 0.5f) * 0.5f) * bh;
        s += bilin(offmap, N_PAD, bin * 2 + c, gy, gx);
      }
    s *= 0.25f;
    s *= (c == 0 ? rw : rh) * LAMDA;
    offs[bin][c] = s;
  }
  __syncthreads();

  // stage 2: deformable PS-RoI pooling of the features
  if (tid < KPOOL * KPOOL * C_OUT) {
    int bin = tid / C_OUT, c = tid % C_OUT;
    int bi = bin / KPOOL, bj = bin % KPOOL;
    float dx = offs[bin][0] / FSTRIDE;
    float dy = offs[bin][1] / FSTRIDE;
    float s = 0.0f;
#pragma unroll
    for (int sy = 0; sy < 2; ++sy)
#pragma unroll
      for (int sx = 0; sx < 2; ++sx) {
        float gx = fx1 + ((float)bj + ((float)sx + 0.5f) * 0.5f) * bw + dx;
        float gy = fy1 + ((float)bi + ((float)sy + 0.5f) * 0.5f) * bh + dy;
        s += bilin(feat, C_IN, bin * C_OUT + c, gy, gx);
      }
    out[(size_t)r * (KPOOL * KPOOL * C_OUT) + tid] = s * 0.25f;
  }
}

// ---------------------------------------------------------------------------
extern "C" void kernel_launch(void* const* d_in, const int* in_sizes, int n_in,
                              void* d_out, int out_size, void* d_ws, size_t ws_size,
                              hipStream_t stream) {
  const float* feat  = (const float*)d_in[0];   // [1,200,200,490]
  const float* rois  = (const float*)d_in[1];   // [512,5]
  const float* convw = (const float*)d_in[2];   // [3,3,490,98]

  char* ws = (char*)d_ws;
  unsigned short* featb = (unsigned short*)ws;                        // 40,960,000 B
  unsigned short* wb    = (unsigned short*)(ws + 40960000);           //  1,032,192 B
  float*          offm  = (float*)(ws + 40960000 + 1032192);          // 17,920,000 B
  // total workspace: 59,912,192 bytes

  const int n1 = N_PIX * C_IN_PAD;
  pack_feat<<<(n1 + 255) / 256, 256, 0, stream>>>(feat, featb);
  const int n2 = N_PAD * K_TOTAL;
  pack_w<<<(n2 + 255) / 256, 256, 0, stream>>>(convw, wb);

  // 2500 tiles / 2 per wave = 1250 jobs = 125 blocks x 10 waves (exact)
  conv_wmma<<<125, 320, 0, stream>>>(featb, wb, offm);

  roi_pool<<<N_ROIS, 512, 0, stream>>>(feat, offm, rois, (float*)d_out);
}
